// CascadedNN_37855841747230
// MI455X (gfx1250) — compile-verified, matrix operand order checked by reading, not run on hardware
//
#include <hip/hip_runtime.h>
#include <hip/hip_bf16.h>

// Problem constants (from reference)
#define B_ROWS 65536
#define D_IN   256
#define H1_N   512
#define H2_N   256
#define T_OUT  64

typedef __bf16 bf16;
typedef __attribute__((ext_vector_type(16))) __bf16 v16bf;
typedef __attribute__((ext_vector_type(8)))  __bf16 v8bf;
typedef __attribute__((ext_vector_type(8)))  float  v8f;
typedef __attribute__((ext_vector_type(4)))  unsigned int u32x4;
typedef __attribute__((ext_vector_type(8)))  unsigned int u32x8;

#define M_TILE   64        // rows per block
#define NTHREADS 128       // 4 waves (wave32)
#define XP   264           // x/f2 LDS pitch (K=256, +8 pad)
#define F1P  520           // f1  LDS pitch (K=512, +8 pad)
#define WP1  264           // W-tile pitch for K=256
#define WP2  520           // W-tile pitch for K=512
#define SCANP 65           // scan buffer pitch (f32, +1 pad)

// prepacked weight image sizes (bf16 elements)
#define PK1_E (8 * 64 * WP1)   // 135168 : W1T, 8 tiles of 64x264
#define PK2_E (8 * 32 * WP2)   // 133120 : W2T, 8 tiles of 32x520
#define PKH_E (64 * WP1)       //  16896 : head, 1 tile of 64x264

struct __align__(16) SMem {
    bf16 wstage[64 * WP1];            // 33792 B: W1T 64x264 | W2T 32x520 | headT 64x264
    union {
        bf16  f1[M_TILE * F1P];       // 66560 B
        float scan[M_TILE * SCANP];   // 16640 B (overlaps f1)
    };
    bf16 xf2[M_TILE * XP];            // 33792 B: x tile -> f2
    float b1s[H1_N];
    float b2s[H2_N];
    float bcs[T_OUT - 1];
    float wps[T_OUT - 1];
};                                     // ~134.5 KB

__device__ __forceinline__ float sigmoidf_(float x) { return 1.0f / (1.0f + __expf(-x)); }

__device__ __forceinline__ v16bf ld_frag(const bf16* p0) {
    union { v16bf v; v8bf h[2]; } u;
    u.h[0] = *(const v8bf*)(p0);
    u.h[1] = *(const v8bf*)(p0 + 16);
    return u.v;
}

// ---- Tensor Data Mover: 1-D contiguous DMA, global -> LDS (ISA ch.8 D# layout) ----
// nelem8 = number of 8-byte elements. All inputs wave-uniform.
__device__ __forceinline__ void tdm_load_1d(unsigned int lds_off, const void* gptr,
                                            unsigned int nelem8) {
    unsigned long long ga = (unsigned long long)(uintptr_t)gptr;
    u32x4 g0;
    g0[0] = 0x1u;                                               // count=1, user mode
    g0[1] = lds_off;                                            // lds_addr (bits 63:32)
    g0[2] = (unsigned int)ga;                                   // global_addr lo
    g0[3] = (unsigned int)((ga >> 32) & 0x1FFFFFFu) | 0x80000000u; // addr hi | type=2
    u32x8 g1;
    g1[0] = 0x3u << 16;                                         // data_size = 8B
    g1[1] = (nelem8 & 0xFFFFu) << 16;                           // tensor_dim0[15:0]
    g1[2] = ((nelem8 >> 16) & 0xFFFFu) | (1u << 16);            // tensor_dim0[31:16] | tensor_dim1=1
    g1[3] = (nelem8 & 0xFFFFu) << 16;                           // tile_dim0
    g1[4] = 1u;                                                 // tile_dim1=1, tile_dim2=0
    g1[5] = nelem8;                                             // tensor_dim0_stride lo32
    g1[6] = 0u;
    g1[7] = 0u;
    asm volatile("tensor_load_to_lds %0, %1" :: "s"(g0), "s"(g1) : "memory");
}

// ---- Async global -> LDS, 16B per lane (ASYNCcnt path) ----
__device__ __forceinline__ void async_cp16(unsigned int lds_off, const void* g) {
    asm volatile("global_load_async_to_lds_b128 %0, %1, off"
                 :: "v"(lds_off), "v"(g) : "memory");
}
__device__ __forceinline__ void wait_async0() {
    asm volatile("s_wait_asynccnt 0x0" ::: "memory");
}

// ================= one-time weight prepack: fp32 -> bf16, transposed, padded =================
__global__ __launch_bounds__(256)
void prepack_weights(const float* __restrict__ W1, const float* __restrict__ W2,
                     const float* __restrict__ Wf, const float* __restrict__ Wc,
                     bf16* __restrict__ ws)
{
    const int stride = gridDim.x * blockDim.x;
    const int gid = blockIdx.x * blockDim.x + threadIdx.x;
    for (int i = gid; i < PK1_E; i += stride) {                 // W1T tiles (64 cols x 256 K)
        int nt = i / (64 * WP1), rem = i - nt * 64 * WP1;
        int n = rem / WP1, k = rem - n * WP1;
        float v = (k < D_IN) ? W1[(size_t)k * H1_N + nt * 64 + n] : 0.f;
        ws[i] = (bf16)v;
    }
    bf16* ws2 = ws + PK1_E;
    for (int i = gid; i < PK2_E; i += stride) {                 // W2T tiles (32 cols x 512 K)
        int nt = i / (32 * WP2), rem = i - nt * 32 * WP2;
        int n = rem / WP2, k = rem - n * WP2;
        float v = (k < H1_N) ? W2[(size_t)k * H2_N + nt * 32 + n] : 0.f;
        ws2[i] = (bf16)v;
    }
    bf16* wsh = ws2 + PK2_E;
    for (int i = gid; i < PKH_E; i += stride) {                 // head tile: col0=Wf, 1..63=Wc rows
        int n = i / WP1, k = i - n * WP1;
        float v = 0.f;
        if (k < H2_N) v = (n == 0) ? Wf[k] : Wc[(size_t)(n - 1) * (H2_N + 1) + k];
        wsh[i] = (bf16)v;
    }
}

// ================= fused forward kernel =================
__global__ __launch_bounds__(NTHREADS)
void cascaded_fused_wmma(const float* __restrict__ x,  const float* __restrict__ W1,
                         const float* __restrict__ b1, const float* __restrict__ W2,
                         const float* __restrict__ b2, const float* __restrict__ Wf,
                         const float* __restrict__ bfp, const float* __restrict__ Wc,
                         const float* __restrict__ bc,  float* __restrict__ out,
                         const bf16* __restrict__ wpack)
{
    __shared__ SMem sm;
    const int tid     = threadIdx.x;
    const int lane    = tid & 31;
    const int wv      = tid >> 5;
    const int rowbase = blockIdx.x * M_TILE;

    const int frow = lane & 15;        // row (A) / col (B,D) within 16x16 tile
    const int kofs = (lane >> 4) << 3; // K sub-offset per ISA 16-bit layout
    const int mb   = (lane >> 4) * 8;  // D-matrix row base per lane half

    const int wvs = __builtin_amdgcn_readfirstlane(wv);                 // scalar wave id
    const unsigned int wbase = (unsigned int)(uintptr_t)(void*)sm.wstage; // LDS byte offset
    const char* pk1 = (const char*)wpack;
    const char* pk2 = pk1 + PK1_E * 2;
    const char* pkh = pk2 + PK2_E * 2;

    // ---- Stage biases, cascade constants, and x tile (fp32 -> bf16) ----
    for (int i = tid; i < H1_N; i += NTHREADS) sm.b1s[i] = b1[i];
    for (int i = tid; i < H2_N; i += NTHREADS) sm.b2s[i] = b2[i];
    for (int i = tid; i < T_OUT - 1; i += NTHREADS) {
        sm.bcs[i] = bc[i];
        sm.wps[i] = Wc[(size_t)i * (H2_N + 1) + H2_N];   // w_prev column
    }
    for (int i = tid; i < M_TILE * D_IN; i += NTHREADS) {
        int r = i >> 8, k = i & 255;
        sm.xf2[r * XP + k] = (bf16)x[(size_t)(rowbase + r) * D_IN + k];
    }
    __syncthreads();

    // ================= GEMM1: f1 = relu(x @ W1 + b1)  (K=256, N=512) =================
    for (int nt = 0; nt < H1_N / 64; ++nt) {
        const int n0 = nt * 64;
        if (wpack) {   // TDM: each wave DMAs its quarter of the 33792B pre-imaged tile
            tdm_load_1d(wbase + wvs * 8448u, pk1 + (size_t)nt * 33792 + wvs * 8448, 1056u);
            __builtin_amdgcn_s_wait_tensorcnt(0);
        } else {       // fallback: load fp32 + convert
            for (int i = tid; i < 64 * D_IN; i += NTHREADS) {
                int n = i & 63, k = i >> 6;
                sm.wstage[n * WP1 + k] = (bf16)W1[(size_t)k * H1_N + n0 + n];
            }
        }
        __syncthreads();
        const bf16* arow = sm.xf2 + (wv * 16 + frow) * XP;
        #pragma unroll
        for (int ns = 0; ns < 4; ++ns) {
            const bf16* brow = sm.wstage + (ns * 16 + frow) * WP1;
            v8f acc = {};
            #pragma unroll
            for (int kc = 0; kc < D_IN / 32; ++kc) {
                v16bf a = ld_frag(arow + kc * 32 + kofs);
                v16bf b = ld_frag(brow + kc * 32 + kofs);
                acc = __builtin_amdgcn_wmma_f32_16x16x32_bf16(false, a, false, b,
                                                              (short)0, acc, false, false);
            }
            const int  ncol = n0 + ns * 16 + frow;
            const float bia = sm.b1s[ncol];
            #pragma unroll
            for (int r = 0; r < 8; ++r) {
                float v = acc[r] + bia; v = v > 0.f ? v : 0.f;
                sm.f1[(wv * 16 + mb + r) * F1P + ncol] = (bf16)v;
            }
        }
        __syncthreads();
    }

    // ================= GEMM2: f2 = relu(f1 @ W2 + b2)  (K=512, N=256) =================
    for (int nt = 0; nt < H2_N / 32; ++nt) {
        const int n0 = nt * 32;
        if (wpack) {
            tdm_load_1d(wbase + wvs * 8320u, pk2 + (size_t)nt * 33280 + wvs * 8320, 1040u);
            __builtin_amdgcn_s_wait_tensorcnt(0);
        } else {
            for (int i = tid; i < 32 * H1_N; i += NTHREADS) {
                int n = i & 31, k = i >> 5;
                sm.wstage[n * WP2 + k] = (bf16)W2[(size_t)k * H2_N + n0 + n];
            }
        }
        __syncthreads();
        const bf16* arow = sm.f1 + (wv * 16 + frow) * F1P;
        #pragma unroll
        for (int ns = 0; ns < 2; ++ns) {
            const bf16* brow = sm.wstage + (ns * 16 + frow) * WP2;
            v8f acc = {};
            #pragma unroll
            for (int kc = 0; kc < H1_N / 32; ++kc) {
                v16bf a = ld_frag(arow + kc * 32 + kofs);
                v16bf b = ld_frag(brow + kc * 32 + kofs);
                acc = __builtin_amdgcn_wmma_f32_16x16x32_bf16(false, a, false, b,
                                                              (short)0, acc, false, false);
            }
            const int  ncol = n0 + ns * 16 + frow;
            const float bia = sm.b2s[ncol];
            #pragma unroll
            for (int r = 0; r < 8; ++r) {
                float v = acc[r] + bia; v = v > 0.f ? v : 0.f;
                sm.xf2[(wv * 16 + mb + r) * XP + ncol] = (bf16)v;   // f2 reuses x region
            }
        }
        __syncthreads();
    }

    // ====== GEMM3 (head): col0 = f2@Wf ; cols 1..63 = f2 @ Wc[:, :256]^T  (K=256, N=64) ======
    if (wpack) {   // ASYNCcnt path: 16B-per-lane async copies of the pre-imaged head tile
        for (int i = tid; i < 2112; i += NTHREADS)
            async_cp16(wbase + i * 16u, pkh + i * 16);
        wait_async0();
    } else {
        for (int i = tid; i < 64 * H2_N; i += NTHREADS) {
            int n = i & 63, k = i >> 6;
            float w = (n == 0) ? Wf[k] : Wc[(size_t)(n - 1) * (H2_N + 1) + k];
            sm.wstage[n * WP1 + k] = (bf16)w;
        }
    }
    __syncthreads();
    {
        const bf16* arow = sm.xf2 + (wv * 16 + frow) * XP;   // f2
        #pragma unroll
        for (int ns = 0; ns < 4; ++ns) {
            const bf16* brow = sm.wstage + (ns * 16 + frow) * WP1;
            v8f acc = {};
            #pragma unroll
            for (int kc = 0; kc < H2_N / 32; ++kc) {
                v16bf a = ld_frag(arow + kc * 32 + kofs);
                v16bf b = ld_frag(brow + kc * 32 + kofs);
                acc = __builtin_amdgcn_wmma_f32_16x16x32_bf16(false, a, false, b,
                                                              (short)0, acc, false, false);
            }
            const int ncol = ns * 16 + frow;
            #pragma unroll
            for (int r = 0; r < 8; ++r)
                sm.scan[(wv * 16 + mb + r) * SCANP + ncol] = acc[r];  // pre-activation
        }
    }
    __syncthreads();

    // ====== Sequential cascade scan: one thread per row, in-place in LDS ======
    if (tid < M_TILE) {
        float* srow = sm.scan + tid * SCANP;
        float prev = sigmoidf_(srow[0] + bfp[0]);
        srow[0] = prev;
        #pragma unroll 1
        for (int t = 0; t < T_OUT - 1; ++t) {
            float s = sigmoidf_(srow[t + 1] + sm.bcs[t] + sm.wps[t] * prev);
            prev = s > prev ? s : prev;
            srow[t + 1] = prev;
        }
    }
    __syncthreads();

    // ====== Coalesced output store (B x 64, f32) ======
    for (int i = tid; i < M_TILE * T_OUT; i += NTHREADS) {
        int r = i >> 6, c = i & 63;
        out[(size_t)(rowbase + r) * T_OUT + c] = sm.scan[r * SCANP + c];
    }
}

extern "C" void kernel_launch(void* const* d_in, const int* in_sizes, int n_in,
                              void* d_out, int out_size, void* d_ws, size_t ws_size,
                              hipStream_t stream) {
    const float* x  = (const float*)d_in[0];
    const float* W1 = (const float*)d_in[1];
    const float* b1 = (const float*)d_in[2];
    const float* W2 = (const float*)d_in[3];
    const float* b2 = (const float*)d_in[4];
    const float* Wf = (const float*)d_in[5];
    const float* bf = (const float*)d_in[6];
    const float* Wc = (const float*)d_in[7];
    const float* bc = (const float*)d_in[8];
    float* out = (float*)d_out;

    const size_t WS_NEEDED = (size_t)(PK1_E + PK2_E + PKH_E) * sizeof(bf16); // ~557 KB
    bf16* wpack = nullptr;
    if (ws_size >= WS_NEEDED) {
        wpack = (bf16*)d_ws;
        hipLaunchKernelGGL(prepack_weights, dim3(512), dim3(256), 0, stream,
                           W1, W2, Wf, Wc, wpack);
    }
    dim3 grid(B_ROWS / M_TILE);   // 1024 blocks
    dim3 block(NTHREADS);         // 128 threads = 4 wave32
    hipLaunchKernelGGL(cascaded_fused_wmma, grid, block, 0, stream,
                       x, W1, b1, W2, b2, Wf, bf, Wc, bc, out, wpack);
}
